// SimpleAttentionModel_39943195853086
// MI455X (gfx1250) — matrix operand order repaired and checked by reading, not compiled, not run on hardware
//
#include <hip/hip_runtime.h>
#include <stdint.h>

typedef __attribute__((ext_vector_type(16))) _Float16 v16h;
typedef __attribute__((ext_vector_type(8)))  float    v8f;
typedef __attribute__((ext_vector_type(4)))  unsigned int u32x4;
typedef __attribute__((ext_vector_type(8)))  int      i32x8;
typedef __attribute__((ext_vector_type(4)))  int      i32x4;

namespace {
constexpr int kB = 2;
constexpr int kH = 16;
constexpr int kS = 2048;
constexpr int kD = 64;
constexpr int kQTile = 16;   // query rows per wave (WMMA M)
constexpr int kKTile = 32;   // keys per main-loop iteration (WMMA K for P@V)
constexpr int kElems = kB * kH * kS * kD;   // 4,194,304 per tensor
}

#if __has_builtin(__builtin_amdgcn_tensor_load_to_lds) && __has_builtin(__builtin_amdgcn_s_wait_tensorcnt)
#define USE_TDM 1
#else
#define USE_TDM 0
#endif

// ---------------------------------------------------------------------------
// Pre-pass: K -> f16 (same layout), V -> f16 transposed to [bh][D][S]
// ---------------------------------------------------------------------------
__global__ __launch_bounds__(256)
void cvt_k_f16(const float* __restrict__ K, _Float16* __restrict__ Kh) {
  const int i = blockIdx.x * 256 + threadIdx.x;
  Kh[i] = (_Float16)K[i];
}

__global__ __launch_bounds__(256)
void cvt_v_f16_t(const float* __restrict__ V, _Float16* __restrict__ Vth) {
  const int i  = blockIdx.x * 256 + threadIdx.x;   // flat over [bh][s][d]
  const int d  = i % kD;
  const int s  = (i / kD) % kS;
  const int bh = i / (kD * kS);
  Vth[((size_t)bh * kD + d) * kS + s] = (_Float16)V[i];
}

// ---------------------------------------------------------------------------
// TDM: async copy of one wave-uniform 32x64 f16 K tile into LDS.
// D# per CDNA5 ISA sect. 8.3/8.4 (2D tensor, 2-byte elems).
// amdgpu-toolchain (clang-23) 6-arg builtin form:
//   (u32x4 g0, i32x8 g1, i32x4 g2, i32x4 g3, i32x8 gx, i32 cpol)
// ---------------------------------------------------------------------------
#if USE_TDM
__device__ __forceinline__
void tdm_load_k_tile(const _Float16* gsrc, uint32_t lds_off) {
  const uint64_t ga = (uint64_t)(uintptr_t)gsrc;
  u32x4 g0;
  g0[0] = 1u;                                            // count=1, user mode
  g0[1] = lds_off;                                       // LDS byte address
  g0[2] = (uint32_t)ga;                                  // global_addr[31:0]
  g0[3] = ((uint32_t)(ga >> 32) & 0x01FFFFFFu) | (2u << 30); // addr[56:32], type=2
  i32x8 g1 = (i32x8)0;
  g1[0] = 1 << 16;                 // data_size = 1 -> 2 bytes; wg_mask = 0
  g1[1] = kD << 16;                // tensor_dim0 = 64      (bits 79:48 lo half)
  g1[2] = (int)((unsigned)kS << 16); // tensor_dim1 = 2048  (bits 111:80 lo half)
  g1[3] = kD << 16;                // tile_dim0 = 64        (bits 127:112)
  g1[4] = kKTile;                  // tile_dim1 = 32; tile_dim2 = 0
  g1[5] = kD;                      // tensor_dim0_stride = 64 elems
  // g1[6], g1[7] = 0 (stride hi bits / dim1 stride unused for 2D tile)
  const i32x4 z4 = (i32x4)0;
  const i32x8 z8 = (i32x8)0;
  __builtin_amdgcn_tensor_load_to_lds(g0, g1, z4, z4, z8, 0);
}
#endif

// ---------------------------------------------------------------------------
// Main flash-attention kernel: one wave32 per 16-row query tile.
// f16 WMMA multiplies, f32 accumulation, online softmax.
// K tiles arrive via TDM double-buffer in LDS; V read from transposed f16.
// ---------------------------------------------------------------------------
__global__ __launch_bounds__(32)
void fa_fwd_wmma_tdm(const float* __restrict__ Qg,
                     const _Float16* __restrict__ Kh,
                     const _Float16* __restrict__ Vth,
                     const float* __restrict__ dsqrt,
                     float* __restrict__ Og)
{
  __shared__ _Float16 kTile[2][kKTile * kD];   // 2 x 4 KB TDM double buffer
  __shared__ _Float16 pTile[kQTile * kKTile];  // 1 KB P re-layout tile

  const int lane = threadIdx.x;   // 0..31 (wave32)
  const int lo   = lane & 15;
  const int hi   = lane >> 4;

  const int qTilesPerHead = kS / kQTile;
  const int bh = blockIdx.x / qTilesPerHead;
  const int q0 = (blockIdx.x % qTilesPerHead) * kQTile;

  const float scale = 1.0f / dsqrt[0];

  const float*    qBase  = Qg  + ((size_t)bh * kS + q0) * kD;
  const _Float16* khBase = Kh  + (size_t)bh * kS * kD;
  const _Float16* vtBase = Vth + (size_t)bh * kD * kS;
  float*          oBase  = Og  + ((size_t)bh * kS + q0) * kD;

  // ---- Q tile as two 16x32 f16 A-fragments, pre-scaled by 1/sqrt(D) ----
  v16h aQ[2];
  {
    const float* qRow = qBase + (size_t)lo * kD;
    #pragma unroll
    for (int c = 0; c < 2; ++c)
      #pragma unroll
      for (int g = 0; g < 2; ++g)
        #pragma unroll
        for (int t = 0; t < 8; ++t)
          aQ[c][8 * g + t] = (_Float16)(qRow[32 * c + 16 * hi + 8 * g + t] * scale);
  }

  v8f acc[4] = {};
  float m[8], l[8];
  #pragma unroll
  for (int j = 0; j < 8; ++j) { m[j] = -3.0e38f; l[j] = 0.0f; }

#if USE_TDM
  const uint32_t ldsK0 = (uint32_t)(uintptr_t)&kTile[0][0];
  const uint32_t ldsK1 = (uint32_t)(uintptr_t)&kTile[1][0];
  tdm_load_k_tile(khBase, ldsK0);                 // prime buffer 0
#endif

  for (int k0 = 0; k0 < kS; k0 += kKTile) {
    const int cur = (k0 / kKTile) & 1;

#if USE_TDM
    if (k0 + kKTile < kS) {
      tdm_load_k_tile(khBase + (size_t)(k0 + kKTile) * kD, cur ? ldsK0 : ldsK1);
      __builtin_amdgcn_s_wait_tensorcnt(1);       // in-order TDM: tile k0 landed
    } else {
      __builtin_amdgcn_s_wait_tensorcnt(0);
    }
    const _Float16* kb = &kTile[cur][0];
    v16h bK[2][2];
    #pragma unroll
    for (int kn = 0; kn < 2; ++kn)
      #pragma unroll
      for (int c = 0; c < 2; ++c)
        bK[kn][c] = *(const v16h*)(kb + (16 * kn + lo) * kD + 32 * c + 16 * hi);
#else
    v16h bK[2][2];
    #pragma unroll
    for (int kn = 0; kn < 2; ++kn) {
      const _Float16* kRow = khBase + (size_t)(k0 + 16 * kn + lo) * kD;
      #pragma unroll
      for (int c = 0; c < 2; ++c)
        bK[kn][c] = *(const v16h*)(kRow + 32 * c + 16 * hi);
    }
#endif

    // ---- scores S = (Q/sqrt(d)) @ K^T for a 16x32 block ----
    v8f s0 = {}, s1 = {};
    s0 = __builtin_amdgcn_wmma_f32_16x16x32_f16(false, aQ[0], false, bK[0][0], (short)0, s0, false, false);
    s0 = __builtin_amdgcn_wmma_f32_16x16x32_f16(false, aQ[1], false, bK[0][1], (short)0, s0, false, false);
    s1 = __builtin_amdgcn_wmma_f32_16x16x32_f16(false, aQ[0], false, bK[1][0], (short)0, s1, false, false);
    s1 = __builtin_amdgcn_wmma_f32_16x16x32_f16(false, aQ[1], false, bK[1][1], (short)0, s1, false, false);

    // ---- online softmax over this 16x32 block ----
    float p0[8], p1[8];
    #pragma unroll
    for (int j = 0; j < 8; ++j) {
      float rm = fmaxf(s0[j], s1[j]);
      rm = fmaxf(rm, __shfl_xor(rm, 1, 16));
      rm = fmaxf(rm, __shfl_xor(rm, 2, 16));
      rm = fmaxf(rm, __shfl_xor(rm, 4, 16));
      rm = fmaxf(rm, __shfl_xor(rm, 8, 16));
      const float mn    = fmaxf(m[j], rm);
      const float alpha = __expf(m[j] - mn);
      m[j] = mn;
      p0[j] = __expf(s0[j] - mn);
      p1[j] = __expf(s1[j] - mn);
      float rs = p0[j] + p1[j];
      rs += __shfl_xor(rs, 1, 16);
      rs += __shfl_xor(rs, 2, 16);
      rs += __shfl_xor(rs, 4, 16);
      rs += __shfl_xor(rs, 8, 16);
      l[j] = l[j] * alpha + rs;
      #pragma unroll
      for (int dc = 0; dc < 4; ++dc) acc[dc][j] *= alpha;
    }

    // ---- stage P (f16) through LDS: D-frag layout -> A-frag layout ----
    #pragma unroll
    for (int j = 0; j < 8; ++j) {
      pTile[(j + 8 * hi) * kKTile + lo]      = (_Float16)p0[j];
      pTile[(j + 8 * hi) * kKTile + 16 + lo] = (_Float16)p1[j];
    }
    __syncthreads();   // single-wave WG: just forces the ds waits
    v16h aP;
    #pragma unroll
    for (int g = 0; g < 2; ++g)
      #pragma unroll
      for (int t = 0; t < 8; ++t)
        aP[8 * g + t] = pTile[lo * kKTile + 16 * hi + 8 * g + t];
    __syncthreads();

    // ---- O += P @ V : contiguous 16-half loads from transposed f16 V ----
    #pragma unroll
    for (int dc = 0; dc < 4; ++dc) {
      const v16h bV = *(const v16h*)(vtBase + ((size_t)(16 * dc + lo)) * kS + k0 + 16 * hi);
      acc[dc] = __builtin_amdgcn_wmma_f32_16x16x32_f16(false, aP, false, bV, (short)0, acc[dc], false, false);
    }

    if (k0 + kKTile < kS)   // global_prefetch_b8 for next V columns
      __builtin_prefetch(vtBase + (size_t)lo * kS + k0 + kKTile, 0, 1);
  }

  // ---- normalize by row sums and store ----
  #pragma unroll
  for (int j = 0; j < 8; ++j) {
    const float inv = 1.0f / l[j];
    float* oRow = oBase + (size_t)(j + 8 * hi) * kD;
    #pragma unroll
    for (int dc = 0; dc < 4; ++dc)
      oRow[16 * dc + lo] = acc[dc][j] * inv;
  }
}

// ---------------------------------------------------------------------------
// Fallback (ws too small): round-1 all-f32-input kernel.
// ---------------------------------------------------------------------------
__global__ __launch_bounds__(32)
void fa_fwd_wmma_f32in(const float* __restrict__ Qg,
                       const float* __restrict__ Kg,
                       const float* __restrict__ Vg,
                       const float* __restrict__ dsqrt,
                       float* __restrict__ Og)
{
  __shared__ _Float16 pTile[kQTile * kKTile];
  const int lane = threadIdx.x, lo = lane & 15, hi = lane >> 4;
  const int qTilesPerHead = kS / kQTile;
  const int bh = blockIdx.x / qTilesPerHead;
  const int q0 = (blockIdx.x % qTilesPerHead) * kQTile;
  const float scale = 1.0f / dsqrt[0];
  const float* qBase = Qg + ((size_t)bh * kS + q0) * kD;
  const float* kBase = Kg + (size_t)bh * kS * kD;
  const float* vBase = Vg + (size_t)bh * kS * kD;
  float*       oBase = Og + ((size_t)bh * kS + q0) * kD;

  v16h aQ[2];
  {
    const float* qRow = qBase + (size_t)lo * kD;
    #pragma unroll
    for (int c = 0; c < 2; ++c)
      #pragma unroll
      for (int g = 0; g < 2; ++g)
        #pragma unroll
        for (int t = 0; t < 8; ++t)
          aQ[c][8 * g + t] = (_Float16)(qRow[32 * c + 16 * hi + 8 * g + t] * scale);
  }
  v8f acc[4] = {};
  float m[8], l[8];
  #pragma unroll
  for (int j = 0; j < 8; ++j) { m[j] = -3.0e38f; l[j] = 0.0f; }

  for (int k0 = 0; k0 < kS; k0 += kKTile) {
    v16h bK[2][2];
    #pragma unroll
    for (int kn = 0; kn < 2; ++kn) {
      const float* kRow = kBase + (size_t)(k0 + 16 * kn + lo) * kD;
      #pragma unroll
      for (int c = 0; c < 2; ++c)
        #pragma unroll
        for (int t = 0; t < 16; ++t)
          bK[kn][c][t] = (_Float16)kRow[32 * c + 16 * hi + t];
    }
    v8f s0 = {}, s1 = {};
    s0 = __builtin_amdgcn_wmma_f32_16x16x32_f16(false, aQ[0], false, bK[0][0], (short)0, s0, false, false);
    s0 = __builtin_amdgcn_wmma_f32_16x16x32_f16(false, aQ[1], false, bK[0][1], (short)0, s0, false, false);
    s1 = __builtin_amdgcn_wmma_f32_16x16x32_f16(false, aQ[0], false, bK[1][0], (short)0, s1, false, false);
    s1 = __builtin_amdgcn_wmma_f32_16x16x32_f16(false, aQ[1], false, bK[1][1], (short)0, s1, false, false);

    float p0[8], p1[8];
    #pragma unroll
    for (int j = 0; j < 8; ++j) {
      float rm = fmaxf(s0[j], s1[j]);
      rm = fmaxf(rm, __shfl_xor(rm, 1, 16));
      rm = fmaxf(rm, __shfl_xor(rm, 2, 16));
      rm = fmaxf(rm, __shfl_xor(rm, 4, 16));
      rm = fmaxf(rm, __shfl_xor(rm, 8, 16));
      const float mn    = fmaxf(m[j], rm);
      const float alpha = __expf(m[j] - mn);
      m[j] = mn;
      p0[j] = __expf(s0[j] - mn);
      p1[j] = __expf(s1[j] - mn);
      float rs = p0[j] + p1[j];
      rs += __shfl_xor(rs, 1, 16);
      rs += __shfl_xor(rs, 2, 16);
      rs += __shfl_xor(rs, 4, 16);
      rs += __shfl_xor(rs, 8, 16);
      l[j] = l[j] * alpha + rs;
      #pragma unroll
      for (int dc = 0; dc < 4; ++dc) acc[dc][j] *= alpha;
    }

    #pragma unroll
    for (int j = 0; j < 8; ++j) {
      pTile[(j + 8 * hi) * kKTile + lo]      = (_Float16)p0[j];
      pTile[(j + 8 * hi) * kKTile + 16 + lo] = (_Float16)p1[j];
    }
    __syncthreads();
    v16h aP;
    #pragma unroll
    for (int g = 0; g < 2; ++g)
      #pragma unroll
      for (int t = 0; t < 8; ++t)
        aP[8 * g + t] = pTile[lo * kKTile + 16 * hi + 8 * g + t];
    __syncthreads();

    #pragma unroll
    for (int dc = 0; dc < 4; ++dc) {
      v16h bV;
      #pragma unroll
      for (int t = 0; t < 16; ++t)
        bV[t] = (_Float16)vBase[(size_t)(k0 + 16 * hi + t) * kD + 16 * dc + lo];
      acc[dc] = __builtin_amdgcn_wmma_f32_16x16x32_f16(false, aP, false, bV, (short)0, acc[dc], false, false);
    }
  }

  #pragma unroll
  for (int j = 0; j < 8; ++j) {
    const float inv = 1.0f / l[j];
    float* oRow = oBase + (size_t)(j + 8 * hi) * kD;
    #pragma unroll
    for (int dc = 0; dc < 4; ++dc)
      oRow[16 * dc + lo] = acc[dc][j] * inv;
  }
}

// ---------------------------------------------------------------------------
extern "C" void kernel_launch(void* const* d_in, const int* in_sizes, int n_in,
                              void* d_out, int out_size, void* d_ws, size_t ws_size,
                              hipStream_t stream) {
  const float* Q     = (const float*)d_in[0];
  const float* K     = (const float*)d_in[1];
  const float* V     = (const float*)d_in[2];
  const float* dsqrt = (const float*)d_in[3];
  float* O = (float*)d_out;

  const int mainBlocks = kB * kH * (kS / kQTile);       // 4096 single-wave WGs
  const size_t bytesF16 = (size_t)kElems * 2;           // 8 MB per f16 tensor

  if (ws_size >= 2 * bytesF16) {
    _Float16* Kh  = (_Float16*)d_ws;
    _Float16* Vth = (_Float16*)((char*)d_ws + bytesF16);
    const int cvtBlocks = kElems / 256;                 // exact: 16384
    cvt_k_f16  <<<cvtBlocks, 256, 0, stream>>>(K, Kh);
    cvt_v_f16_t<<<cvtBlocks, 256, 0, stream>>>(V, Vth);
    fa_fwd_wmma_tdm<<<mainBlocks, 32, 0, stream>>>(Q, Kh, Vth, dsqrt, O);
  } else {
    fa_fwd_wmma_f32in<<<mainBlocks, 32, 0, stream>>>(Q, K, V, dsqrt, O);
  }
}
